// SelfAttentiveHawkesProcesses_77249281786309
// MI455X (gfx1250) — matrix operand order repaired and checked by reading, not compiled, not run on hardware
//
#include <hip/hip_runtime.h>
#include <hip/hip_bf16.h>
#include <math.h>

// ---------------------------------------------------------------------------
// CDNA5 (gfx1250) WMMA / TDM types
// ---------------------------------------------------------------------------
typedef __attribute__((ext_vector_type(16))) _Float16 v16h;
typedef __attribute__((ext_vector_type(8)))  _Float16 v8h;
typedef __attribute__((ext_vector_type(8)))  float    v8f;
typedef __attribute__((ext_vector_type(4)))  unsigned int v4u;
typedef __attribute__((ext_vector_type(4)))  int      v4i;
typedef __attribute__((ext_vector_type(8)))  int      v8i;

#define WMMA16x16x32(a, b, c) \
    __builtin_amdgcn_wmma_f32_16x16x32_f16(false, (a), false, (b), (short)0, (c), false, false)
#define CAT16(lo, hi) \
    __builtin_shufflevector((lo), (hi), 0, 1, 2, 3, 4, 5, 6, 7, 8, 9, 10, 11, 12, 13, 14, 15)

// Problem constants
#define BB 2
#define TT 2049
#define SS 2048
#define DM 512
#define HH 8
#define DK 64
#define DFF 2048
#define NTOK (BB * SS)       // 4096
#define EMB 511

enum { ACT_NONE = 0, ACT_GELU = 1, ACT_SP10 = 2 };
enum { OUT_F16 = 0, OUT_F16_QV = 1, OUT_F16_KT = 2, OUT_F32_RES = 3, OUT_F32_RES_F16 = 4, OUT_F32 = 5 };

__device__ __forceinline__ float gelu_f(float x) {
    return 0.5f * x * (1.0f + tanhf(0.7978845608028654f * (x + 0.044715f * x * x * x)));
}
__device__ __forceinline__ float sp10_f(float v) {
    float x = 10.0f * v;
    float sp = fmaxf(x, 0.0f) + log1pf(__expf(-fabsf(x)));
    return sp * 0.1f;
}

// ---------------------------------------------------------------------------
// Tensor Data Mover: async 2D tile load (global, row-major) -> LDS (compact).
// Descriptor per CDNA5 ISA.
//   group0: count=1 | lds_addr | global_addr[56:0] | type=2
//   group1: data_size=2B, tensor_dim0/1, tile_dim0/1, tensor_dim0_stride
//   group2/3: zero (2D tile: tile_dim2..4 unused, no iteration/gather)
// ---------------------------------------------------------------------------
__device__ __forceinline__ void tdm_load_2d(const void* gptr, unsigned lds_off,
                                            unsigned tdim0, unsigned tdim1,
                                            unsigned stride0,  // elements
                                            unsigned tile0, unsigned tile1) {
    unsigned long long ga = (unsigned long long)(size_t)gptr;
    v4u g0;
    g0.x = 1u;                                             // count=1, user descriptor
    g0.y = lds_off;                                        // lds_addr (bytes)
    g0.z = (unsigned)ga;                                   // global_addr[31:0]
    g0.w = (unsigned)((ga >> 32) & 0x01ffffffull) | (2u << 30); // addr[56:32] | type=2
    v8i g1;
    g1[0] = (int)0x00010000u;                              // data_size=1 (2 bytes)
    g1[1] = (int)((tdim0 & 0xffffu) << 16);                // tensor_dim0[15:0]
    g1[2] = (int)((tdim0 >> 16) | ((tdim1 & 0xffffu) << 16)); // dim0[31:16]|dim1[15:0]
    g1[3] = (int)((tdim1 >> 16) | (tile0 << 16));          // dim1[31:16]|tile_dim0
    g1[4] = (int)(tile1 & 0xffffu);                        // tile_dim1 (tile_dim2=0)
    g1[5] = (int)stride0;                                  // tensor_dim0_stride[31:0]
    g1[6] = 0;                                             // stride0[47:32], dim1_stride lo
    g1[7] = 0;                                             // dim1_stride hi
    v4i g2 = {0, 0, 0, 0};
    v4i g3 = {0, 0, 0, 0};
#if defined(__clang_major__) && (__clang_major__ >= 23)
    v8i g4 = {0, 0, 0, 0, 0, 0, 0, 0};
    __builtin_amdgcn_tensor_load_to_lds(g0, g1, g2, g3, g4, 0);
#else
    __builtin_amdgcn_tensor_load_to_lds(g0, g1, g2, g3, 0);
#endif
}

// ---------------------------------------------------------------------------
// fp32 -> f16 weight conversion
// ---------------------------------------------------------------------------
__global__ void cvt_f16_kernel(const float* __restrict__ src, _Float16* __restrict__ dst, int n) {
    int i = blockIdx.x * 256 + threadIdx.x;
    if (i < n) dst[i] = (_Float16)src[i];
}

// ---------------------------------------------------------------------------
// Feature build (dt + embedding) + LayerNorm -> feat (f32) and xn (f16)
// ---------------------------------------------------------------------------
__launch_bounds__(256)
__global__ void feat_ln_kernel(const float* __restrict__ ev,    // (B,T,2)
                               const float* __restrict__ Wemb,  // (128,511)
                               const float* __restrict__ ln_a,
                               const float* __restrict__ ln_b,
                               float* __restrict__ feat,
                               _Float16* __restrict__ xn) {
    __shared__ float row[DM];
    __shared__ float red[256];
    const int tok = blockIdx.x;
    const int b = tok >> 11, s = tok & (SS - 1);
    const int tid = threadIdx.x;

    const int   ty     = (int)ev[((size_t)b * TT + s) * 2 + 1];
    const float t_cur  = ev[((size_t)b * TT + s) * 2 + 0];
    const float t_prev = (s == 0) ? 0.0f : ev[((size_t)b * TT + s - 1) * 2 + 0];

    for (int i = tid; i < DM; i += 256) {
        float x = (i == 0) ? (t_cur - t_prev) : Wemb[(size_t)ty * EMB + (i - 1)];
        row[i] = x;
        feat[(size_t)tok * DM + i] = x;
    }
    __syncthreads();

    red[tid] = row[tid] + row[tid + 256];
    __syncthreads();
    for (int st = 128; st > 0; st >>= 1) {
        if (tid < st) red[tid] += red[tid + st];
        __syncthreads();
    }
    const float mean = red[0] * (1.0f / (float)DM);
    __syncthreads();

    float d0 = row[tid] - mean, d1 = row[tid + 256] - mean;
    red[tid] = d0 * d0 + d1 * d1;
    __syncthreads();
    for (int st = 128; st > 0; st >>= 1) {
        if (tid < st) red[tid] += red[tid + st];
        __syncthreads();
    }
    const float var  = red[0] * (1.0f / (float)(DM - 1));
    const float rstd = 1.0f / (sqrtf(var) + 1e-6f);

    for (int i = tid; i < DM; i += 256) {
        float y = ln_a[i] * (row[i] - mean) * rstd + ln_b[i];
        xn[(size_t)tok * DM + i] = (_Float16)y;
    }
}

// ---------------------------------------------------------------------------
// Standalone LayerNorm (f32 in -> f16 out)
// ---------------------------------------------------------------------------
__launch_bounds__(256)
__global__ void ln_kernel(const float* __restrict__ x,
                          const float* __restrict__ ln_a,
                          const float* __restrict__ ln_b,
                          _Float16* __restrict__ out) {
    __shared__ float row[DM];
    __shared__ float red[256];
    const int tok = blockIdx.x;
    const int tid = threadIdx.x;

    for (int i = tid; i < DM; i += 256) row[i] = x[(size_t)tok * DM + i];
    __syncthreads();

    red[tid] = row[tid] + row[tid + 256];
    __syncthreads();
    for (int st = 128; st > 0; st >>= 1) {
        if (tid < st) red[tid] += red[tid + st];
        __syncthreads();
    }
    const float mean = red[0] * (1.0f / (float)DM);
    __syncthreads();

    float d0 = row[tid] - mean, d1 = row[tid + 256] - mean;
    red[tid] = d0 * d0 + d1 * d1;
    __syncthreads();
    for (int st = 128; st > 0; st >>= 1) {
        if (tid < st) red[tid] += red[tid + st];
        __syncthreads();
    }
    const float var  = red[0] * (1.0f / (float)(DM - 1));
    const float rstd = 1.0f / (sqrtf(var) + 1e-6f);

    for (int i = tid; i < DM; i += 256) {
        float y = ln_a[i] * (row[i] - mean) * rstd + ln_b[i];
        out[(size_t)tok * DM + i] = (_Float16)y;
    }
}

// ---------------------------------------------------------------------------
// Generic WMMA f16 GEMM: C = act(A[M x K] * B[K x N] + bias) (+ residual)
// Block: 256 threads (8 waves), tile 64(M) x 64(N), K step 32.
// A tile staged by the Tensor Data Mover (async DMA, row-major compact into
// LDS) while all threads stage B transposed (N-major) for contiguous-K
// fragments. Every fragment = two ds_load_b128.
// ---------------------------------------------------------------------------
__launch_bounds__(256)
__global__ void gemm_wmma_kernel(const _Float16* __restrict__ A, int lda,
                                 const _Float16* __restrict__ Bw, int ldb,
                                 const float* __restrict__ bias,
                                 const float* __restrict__ res,
                                 float* __restrict__ Cf,
                                 _Float16* __restrict__ Ch,
                                 int Kdim, int N, int act, int mode) {
    __shared__ __attribute__((aligned(16))) _Float16 As[64 * 32];   // [m][k]  (TDM dest)
    __shared__ __attribute__((aligned(16))) _Float16 Bst[64 * 32];  // [n][k] (transposed)

    const int m0 = blockIdx.y * 64;
    const int n0 = blockIdx.x * 64;
    const int tid  = threadIdx.x;
    const int lane = tid & 31;
    const int w    = tid >> 5;
    const int wm   = (w & 3) * 16;
    const int wn   = (w >> 2) * 32;

    v8f acc0 = {0.f, 0.f, 0.f, 0.f, 0.f, 0.f, 0.f, 0.f};
    v8f acc1 = acc0;

    const int br = tid >> 3;         // 0..31
    const int bc = (tid & 7) * 8;    // 0..56

    const int am  = lane & 15;
    const int akb = (lane < 16) ? 0 : 8;    // A-matrix K base (ISA 16-bit A 16x32 layout)
    const int bn  = lane & 15;
    const int bkb = (lane < 16) ? 0 : 16;   // B-matrix K base
    const int ai  = akb >> 3;               // 0 or 1
    const int bi  = bkb >> 3;               // 0 or 2

    const unsigned as_lds = (unsigned)(size_t)As;

    for (int k0 = 0; k0 < Kdim; k0 += 32) {
        // async A-tile DMA: 64 rows x 32 halfs, row-major compact into As
        if (w == 0) {
            tdm_load_2d(&A[(size_t)m0 * lda + k0], as_lds,
                        (unsigned)Kdim, (unsigned)NTOK, (unsigned)lda, 32u, 64u);
        }
        // stage B transposed while the DMA runs
        {
            uint4 btmp = *(const uint4*)&Bw[(size_t)(k0 + br) * ldb + n0 + bc];
            const _Float16* hp = (const _Float16*)&btmp;
#pragma unroll
            for (int i = 0; i < 8; ++i) Bst[(bc + i) * 32 + br] = hp[i];
        }
        if (k0 + 32 < Kdim) {
            __builtin_prefetch(&Bw[(size_t)(k0 + 32 + br) * ldb + n0 + bc], 0, 1);
        }
        if (w == 0) __builtin_amdgcn_s_wait_tensorcnt(0);
        __syncthreads();

        // A fragment: lane holds K [bkb,bkb+16)
        const v8h* ap8 = (const v8h*)(As + (wm + am) * 32);
        v16h af = CAT16(ap8[bi], ap8[bi + 1]);

        // B fragments: lane holds K [bkb,bkb+16) for its column
        const v8h* bp0 = (const v8h*)(Bst + (wn + bn) * 32);
        const v8h* bp1 = (const v8h*)(Bst + (wn + 16 + bn) * 32);
        v16h bf0 = CAT16(bp0[bi], bp0[bi + 1]);
        v16h bf1 = CAT16(bp1[bi], bp1[bi + 1]);

        acc0 = WMMA16x16x32(af, bf0, acc0);
        acc1 = WMMA16x16x32(af, bf1, acc1);
        __syncthreads();
    }

    // Epilogue: C/D layout -> lane l, VGPR r: (M = r + 8*(l>=16), N = l&15)
    const int hi = lane >> 4;
#pragma unroll
    for (int sub = 0; sub < 2; ++sub) {
        v8f acc = sub ? acc1 : acc0;
#pragma unroll
        for (int r = 0; r < 8; ++r) {
            const int m  = wm + r + 8 * hi;
            const int n  = wn + sub * 16 + (lane & 15);
            const int gm = m0 + m;
            const int gn = n0 + n;
            float val = acc[r] + bias[gn];
            if (act == ACT_GELU)      val = gelu_f(val);
            else if (act == ACT_SP10) val = sp10_f(val);
            if (mode == OUT_F32_RES || mode == OUT_F32_RES_F16) val += res[(size_t)gm * N + gn];
            if (mode == OUT_F16) {
                Ch[(size_t)gm * N + gn] = (_Float16)val;
            } else if (mode == OUT_F16_QV) {
                int b = gm >> 11, s = gm & (SS - 1), h = gn >> 6, d = gn & 63;
                Ch[((size_t)((b * HH + h) * SS + s)) * DK + d] = (_Float16)val;
            } else if (mode == OUT_F16_KT) {
                int b = gm >> 11, s = gm & (SS - 1), h = gn >> 6, d = gn & 63;
                Ch[((size_t)((b * HH + h) * DK + d)) * SS + s] = (_Float16)val;
            } else if (mode == OUT_F32_RES) {
                Cf[(size_t)gm * N + gn] = val;
            } else if (mode == OUT_F32_RES_F16) {
                Cf[(size_t)gm * N + gn] = val;
                Ch[(size_t)gm * N + gn] = (_Float16)val;
            } else { // OUT_F32
                Cf[(size_t)gm * N + gn] = val;
            }
        }
    }
}

// ---------------------------------------------------------------------------
// Attention: one wave per (b,h, 16-row s-tile). Causal; scores = exp(q.k)/8,
// then masked softmax (pass 1: online max/sum; pass 2: normalize, write
// p_attn fp32, accumulate P*V via WMMA).
//  q: [b,h,s,d] f16   (A operand, K contiguous)
//  k: [b,h,t,d] f16   (B operand of scores: (k=d, n=t) -> d contiguous)
//  vT:[b,h,d,t] f16   (B operand of P*V:    (k=t, n=d) -> t contiguous)
// All fragments = two b128 loads.
// ---------------------------------------------------------------------------
__launch_bounds__(32)
__global__ void attention_kernel(const _Float16* __restrict__ q,
                                 const _Float16* __restrict__ kk,
                                 const _Float16* __restrict__ vt,
                                 _Float16* __restrict__ ctx,
                                 float* __restrict__ pout) {
    __shared__ __attribute__((aligned(16))) _Float16 Pt[16 * 32];

    const int blk   = blockIdx.x;           // 0..2047
    const int stile = blk & 127;
    const int bh    = blk >> 7;             // b*H + h
    const int s0    = stile * 16;
    const int lane  = threadIdx.x;
    const int hi    = lane >> 4;
    const int am    = lane & 15;
    const int akb   = hi ? 8 : 0;
    const int bn    = lane & 15;
    const int bkb   = hi ? 16 : 0;
    const int ai    = akb >> 3;             // 0 or 1
    const int bi    = bkb >> 3;             // 0 or 2

    const _Float16* qh = q  + (size_t)bh * SS * DK;
    const _Float16* kh = kk + (size_t)bh * SS * DK;
    const _Float16* vh = vt + (size_t)bh * DK * SS;
    float* prow = pout + (size_t)bh * SS * SS;

    // Preload Q fragments (d in [0,32) and [32,64)); row = 8 chunks of v8h
    const v8h* qr = (const v8h*)(qh + (size_t)(s0 + am) * DK);
    v16h aq0 = CAT16(qr[bi], qr[bi + 1]);
    v16h aq1 = CAT16(qr[4 + bi], qr[5 + bi]);

    const int ntt = stile + 1;   // causal: only t-tiles with t0 <= s0+15
    float rmax[8], rsum[8];
#pragma unroll
    for (int r = 0; r < 8; ++r) { rmax[r] = -3.0e38f; rsum[r] = 0.0f; }

    // ---- pass 1: online per-row max & sum of z = exp(q.k)*0.125 ----
    for (int tt = 0; tt < ntt; ++tt) {
        const int t0 = tt * 16;
        const v8h* kr = (const v8h*)(kh + (size_t)(t0 + bn) * DK);
        v16h bk0 = CAT16(kr[bi], kr[bi + 1]);
        v16h bk1 = CAT16(kr[4 + bi], kr[5 + bi]);
        v8f sc = {0.f, 0.f, 0.f, 0.f, 0.f, 0.f, 0.f, 0.f};
        sc = WMMA16x16x32(aq0, bk0, sc);
        sc = WMMA16x16x32(aq1, bk1, sc);
        const int t = t0 + bn;
#pragma unroll
        for (int r = 0; r < 8; ++r) {
            const int s = s0 + r + 8 * hi;
            if (t <= s) {
                float z  = __expf(sc[r]) * 0.125f;
                float nm = fmaxf(rmax[r], z);
                rsum[r]  = rsum[r] * __expf(rmax[r] - nm) + __expf(z - nm);
                rmax[r]  = nm;
            }
        }
    }
    // cross-lane reduce inside each 16-lane group (rows stay within a half)
#pragma unroll
    for (int off = 1; off < 16; off <<= 1) {
#pragma unroll
        for (int r = 0; r < 8; ++r) {
            float om = __shfl_xor(rmax[r], off, 32);
            float os = __shfl_xor(rsum[r], off, 32);
            float nm = fmaxf(rmax[r], om);
            rsum[r]  = rsum[r] * __expf(rmax[r] - nm) + os * __expf(om - nm);
            rmax[r]  = nm;
        }
    }
    float inv[8];
#pragma unroll
    for (int r = 0; r < 8; ++r) inv[r] = 1.0f / rsum[r];

    // ---- pass 2: recompute scores, write p_attn, accumulate P*V ----
    v8f c0 = {0.f, 0.f, 0.f, 0.f, 0.f, 0.f, 0.f, 0.f};
    v8f cacc[4] = {c0, c0, c0, c0};
    const int nch = (ntt + 1) >> 1;
    for (int tc = 0; tc < nch; ++tc) {
#pragma unroll
        for (int hf = 0; hf < 2; ++hf) {
            const int tt = tc * 2 + hf;
            if (tt < ntt) {
                const int t0 = tt * 16;
                const v8h* kr = (const v8h*)(kh + (size_t)(t0 + bn) * DK);
                v16h bk0 = CAT16(kr[bi], kr[bi + 1]);
                v16h bk1 = CAT16(kr[4 + bi], kr[5 + bi]);
                v8f sc = {0.f, 0.f, 0.f, 0.f, 0.f, 0.f, 0.f, 0.f};
                sc = WMMA16x16x32(aq0, bk0, sc);
                sc = WMMA16x16x32(aq1, bk1, sc);
                const int t = t0 + bn;
#pragma unroll
                for (int r = 0; r < 8; ++r) {
                    const int s = s0 + r + 8 * hi;
                    float p = 0.0f;
                    if (t <= s) {
                        float z = __expf(sc[r]) * 0.125f;
                        p = __expf(z - rmax[r]) * inv[r];
                    }
                    prow[(size_t)s * SS + t] = p;
                    Pt[(r + 8 * hi) * 32 + hf * 16 + bn] = (_Float16)p;
                }
            } else {
#pragma unroll
                for (int r = 0; r < 8; ++r)
                    Pt[(r + 8 * hi) * 32 + hf * 16 + bn] = (_Float16)0.0f;
            }
        }
        // P A-fragment from LDS (16x32, K contiguous -> two ds_load_b128)
        const v8h* pr = (const v8h*)(Pt + am * 32);
        v16h ap = CAT16(pr[bi], pr[bi + 1]);

        const int tbase = tc * 32;
#pragma unroll
        for (int dt4 = 0; dt4 < 4; ++dt4) {
            const v8h* vr = (const v8h*)(vh + (size_t)(dt4 * 16 + bn) * SS + tbase + bkb);
            v16h bv_ = CAT16(vr[0], vr[1]);
            cacc[dt4] = WMMA16x16x32(ap, bv_, cacc[dt4]);
        }
    }

    // write ctx as [tok, 512] f16
    {
        const int b = bh >> 3, h = bh & 7;
#pragma unroll
        for (int dt4 = 0; dt4 < 4; ++dt4) {
#pragma unroll
            for (int r = 0; r < 8; ++r) {
                const int s = s0 + r + 8 * hi;
                const int d = dt4 * 16 + bn;
                ctx[((size_t)(b * SS + s)) * DM + h * DK + d] = (_Float16)cacc[dt4][r];
            }
        }
    }

    // zero the masked region of p_attn beyond the last computed tile (b128)
    {
        float4 z4 = {0.f, 0.f, 0.f, 0.f};
        for (int t = ntt * 16 + lane * 4; t < SS; t += 128) {
#pragma unroll
            for (int r = 0; r < 16; ++r)
                *(float4*)&prow[(size_t)(s0 + r) * SS + t] = z4;
        }
    }
}

// ---------------------------------------------------------------------------
// Mean over heads: tmp [tok, 512] -> out [tok, 64]
// ---------------------------------------------------------------------------
__global__ void head_mean_kernel(const float* __restrict__ t, float* __restrict__ o) {
    int i = blockIdx.x * 256 + threadIdx.x;
    if (i < NTOK * DK) {
        int tok = i >> 6, d = i & 63;
        float s = 0.0f;
#pragma unroll
        for (int h = 0; h < HH; ++h) s += t[(size_t)tok * DM + h * DK + d];
        o[i] = s * 0.125f;
    }
}

// ---------------------------------------------------------------------------
// Host orchestration
// ---------------------------------------------------------------------------
extern "C" void kernel_launch(void* const* d_in, const int* in_sizes, int n_in,
                              void* d_out, int out_size, void* d_ws, size_t ws_size,
                              hipStream_t stream) {
    (void)in_sizes; (void)n_in; (void)out_size; (void)ws_size;

    const float* ev   = (const float*)d_in[0];
    const float* Wemb = (const float*)d_in[2];
    const float* ln_a = (const float*)d_in[3];
    const float* ln_b = (const float*)d_in[4];
    const float* Wq = (const float*)d_in[5];  const float* bq = (const float*)d_in[6];
    const float* Wk = (const float*)d_in[7];  const float* bk = (const float*)d_in[8];
    const float* Wv = (const float*)d_in[9];  const float* bv = (const float*)d_in[10];
    const float* Wo = (const float*)d_in[11]; const float* bo = (const float*)d_in[12];
    const float* W1 = (const float*)d_in[13]; const float* b1 = (const float*)d_in[14];
    const float* W2 = (const float*)d_in[15]; const float* b2 = (const float*)d_in[16];
    const float* Wsw = (const float*)d_in[17]; const float* bs = (const float*)d_in[18];
    const float* Wcw = (const float*)d_in[19]; const float* bc = (const float*)d_in[20];
    const float* Wdw = (const float*)d_in[21]; const float* bd = (const float*)d_in[22];

    char* ws = (char*)d_ws;
    size_t off = 0;
    auto ah = [&](size_t n) -> _Float16* {
        _Float16* p = (_Float16*)(ws + off);
        off += ((n * sizeof(_Float16)) + 255) / 256 * 256;
        return p;
    };
    auto af = [&](size_t n) -> float* {
        float* p = (float*)(ws + off);
        off += ((n * sizeof(float)) + 255) / 256 * 256;
        return p;
    };

    const size_t NW = (size_t)DM * DM;       // 262144
    _Float16* wq_h = ah(NW);
    _Float16* wk_h = ah(NW);
    _Float16* wv_h = ah(NW);
    _Float16* wo_h = ah(NW);
    _Float16* ws_h = ah(NW);
    _Float16* wc_h = ah(NW);
    _Float16* wd_h = ah(NW);
    _Float16* w1_h = ah((size_t)DM * DFF);
    _Float16* w2_h = ah((size_t)DFF * DM);
    float*    feat   = af((size_t)NTOK * DM);
    _Float16* xn     = ah((size_t)NTOK * DM);
    _Float16* qbuf   = ah((size_t)NTOK * DM);
    _Float16* kbuf   = ah((size_t)NTOK * DM);
    _Float16* vtbuf  = ah((size_t)NTOK * DM);
    _Float16* ctx_h  = ah((size_t)NTOK * DM);
    float*    feat2  = af((size_t)NTOK * DM);
    _Float16* xn2    = ah((size_t)NTOK * DM);
    _Float16* ffh    = ah((size_t)NTOK * DFF);
    float*    feat3  = af((size_t)NTOK * DM);
    _Float16* feat3h = ah((size_t)NTOK * DM);
    float*    tmp    = af((size_t)NTOK * DM);

    float* out_start = (float*)d_out;                 // (B,S,64)
    float* out_conv  = out_start + (size_t)NTOK * DK;
    float* out_omega = out_conv + (size_t)NTOK * DK;
    float* out_p     = out_omega + (size_t)NTOK * DK; // (B,H,S,S)

    // weight conversions
    {
        const int n = (int)NW;
        int g = (n + 255) / 256;
        cvt_f16_kernel<<<g, 256, 0, stream>>>(Wq, wq_h, n);
        cvt_f16_kernel<<<g, 256, 0, stream>>>(Wk, wk_h, n);
        cvt_f16_kernel<<<g, 256, 0, stream>>>(Wv, wv_h, n);
        cvt_f16_kernel<<<g, 256, 0, stream>>>(Wo, wo_h, n);
        cvt_f16_kernel<<<g, 256, 0, stream>>>(Wsw, ws_h, n);
        cvt_f16_kernel<<<g, 256, 0, stream>>>(Wcw, wc_h, n);
        cvt_f16_kernel<<<g, 256, 0, stream>>>(Wdw, wd_h, n);
        int n2 = DM * DFF;
        int g2 = (n2 + 255) / 256;
        cvt_f16_kernel<<<g2, 256, 0, stream>>>(W1, w1_h, n2);
        cvt_f16_kernel<<<g2, 256, 0, stream>>>(W2, w2_h, n2);
    }

    // features + LN
    feat_ln_kernel<<<NTOK, 256, 0, stream>>>(ev, Wemb, ln_a, ln_b, feat, xn);

    const dim3 g512(DM / 64, NTOK / 64);   // (8, 64)
    const dim3 gff(DFF / 64, NTOK / 64);   // (32, 64)

    // QKV projections: Q,K in [b,h,s,d]; V transposed to [b,h,d,s]
    gemm_wmma_kernel<<<g512, 256, 0, stream>>>(xn, DM, wq_h, DM, bq, nullptr, nullptr, qbuf,  DM, DM, ACT_NONE, OUT_F16_QV);
    gemm_wmma_kernel<<<g512, 256, 0, stream>>>(xn, DM, wk_h, DM, bk, nullptr, nullptr, kbuf,  DM, DM, ACT_NONE, OUT_F16_QV);
    gemm_wmma_kernel<<<g512, 256, 0, stream>>>(xn, DM, wv_h, DM, bv, nullptr, nullptr, vtbuf, DM, DM, ACT_NONE, OUT_F16_KT);

    // attention (writes p_attn + ctx)
    attention_kernel<<<BB * HH * (SS / 16), 32, 0, stream>>>(qbuf, kbuf, vtbuf, ctx_h, out_p);

    // feat2 = feat + ctx@Wo + bo
    gemm_wmma_kernel<<<g512, 256, 0, stream>>>(ctx_h, DM, wo_h, DM, bo, feat, feat2, nullptr, DM, DM, ACT_NONE, OUT_F32_RES);

    // FFN
    ln_kernel<<<NTOK, 256, 0, stream>>>(feat2, ln_a, ln_b, xn2);
    gemm_wmma_kernel<<<gff, 256, 0, stream>>>(xn2, DM, w1_h, DFF, b1, nullptr, nullptr, ffh, DM, DFF, ACT_GELU, OUT_F16);
    gemm_wmma_kernel<<<g512, 256, 0, stream>>>(ffh, DFF, w2_h, DM, b2, feat2, feat3, feat3h, DFF, DM, ACT_NONE, OUT_F32_RES_F16);

    // heads
    const int gm = (NTOK * DK + 255) / 256;
    gemm_wmma_kernel<<<g512, 256, 0, stream>>>(feat3h, DM, ws_h, DM, bs, nullptr, tmp, nullptr, DM, DM, ACT_GELU, OUT_F32);
    head_mean_kernel<<<gm, 256, 0, stream>>>(tmp, out_start);
    gemm_wmma_kernel<<<g512, 256, 0, stream>>>(feat3h, DM, wc_h, DM, bc, nullptr, tmp, nullptr, DM, DM, ACT_GELU, OUT_F32);
    head_mean_kernel<<<gm, 256, 0, stream>>>(tmp, out_conv);
    gemm_wmma_kernel<<<g512, 256, 0, stream>>>(feat3h, DM, wd_h, DM, bd, nullptr, tmp, nullptr, DM, DM, ACT_SP10, OUT_F32);
    head_mean_kernel<<<gm, 256, 0, stream>>>(tmp, out_omega);
}
